// OneTransModel_5248450036152
// MI455X (gfx1250) — compile-verified, matrix operand order checked by reading, not run on hardware
//
#include <hip/hip_runtime.h>
#include <hip/hip_bf16.h>

// ---------------- problem constants ----------------
#define B_   128
#define L_   512
#define H_   256
#define S_   1023          // 2L-1
#define K1_  1536          // 6H
#define N1_  1024          // 4H
#define NROW 65536         // B*L

typedef __attribute__((ext_vector_type(16))) __bf16 v16bf;
typedef __attribute__((ext_vector_type(8)))  float  v8f;
typedef __attribute__((ext_vector_type(4)))  int    v4i;

// CDNA5 async global->LDS copy (ASYNCcnt-tracked, bypasses VGPRs entirely)
#if defined(__has_builtin)
#  if __has_builtin(__builtin_amdgcn_global_load_async_to_lds_b128) && \
      __has_builtin(__builtin_amdgcn_s_wait_asynccnt)
#    define HAVE_ASYNC_LDS 1
#  endif
#endif
#ifndef HAVE_ASYNC_LDS
#  define HAVE_ASYNC_LDS 0
#endif

__device__ __forceinline__ unsigned short f2bf(float f) {
    union { float f; unsigned u; } v; v.f = f;
    unsigned r = v.u + 0x7FFFu + ((v.u >> 16) & 1u);   // round-to-nearest-even
    return (unsigned short)(r >> 16);
}

// ---------------- f32 -> bf16, transposed: dst[n*K+k] = bf16(src[k*N+n]) ------
__global__ void cvt_transpose_bf16(const float* __restrict__ src,
                                   unsigned short* __restrict__ dst,
                                   int K, int N) {
    long i = (long)blockIdx.x * blockDim.x + threadIdx.x;   // output index
    if (i < (long)K * N) {
        int n = (int)(i / K), k = (int)(i % K);
        dst[i] = f2bf(src[(long)k * N + n]);
    }
}

// ---------------- fused embedding gather + LayerNorm -> bf16 X ----------------
__global__ __launch_bounds__(256)
void embed_ln(const int* __restrict__ tok,  const int* __restrict__ post,
              const int* __restrict__ auth, const int* __restrict__ act,
              const int* __restrict__ tg,   const int* __restrict__ gid,
              const float* __restrict__ token_table,
              const float* __restrict__ time_table,
              const float* __restrict__ group_table,
              const float* __restrict__ gamma, const float* __restrict__ beta,
              unsigned short* __restrict__ Xbf)
{
    int r = blockIdx.x, tid = threadIdx.x;
    int tgi = tg[r]; tgi = tgi < 0 ? 0 : (tgi > 128 ? 128 : tgi);
    float x[6];
    x[0] = token_table[(long)tok[r]  * H_ + tid];
    x[1] = token_table[(long)post[r] * H_ + tid];
    x[2] = token_table[(long)auth[r] * H_ + tid];
    x[3] = token_table[(long)act[r]  * H_ + tid];
    x[4] = time_table [(long)tgi     * H_ + tid];
    x[5] = group_table[(long)gid[r]  * H_ + tid];

    __shared__ float red[256];
    float s = x[0]+x[1]+x[2]+x[3]+x[4]+x[5];
    red[tid] = s; __syncthreads();
    #pragma unroll
    for (int off = 128; off > 0; off >>= 1) {
        if (tid < off) red[tid] += red[tid + off];
        __syncthreads();
    }
    float mean = red[0] * (1.0f / K1_);
    __syncthreads();
    float sq = x[0]*x[0]+x[1]*x[1]+x[2]*x[2]+x[3]*x[3]+x[4]*x[4]+x[5]*x[5];
    red[tid] = sq; __syncthreads();
    #pragma unroll
    for (int off = 128; off > 0; off >>= 1) {
        if (tid < off) red[tid] += red[tid + off];
        __syncthreads();
    }
    float var = red[0] * (1.0f / K1_) - mean * mean;
    float inv = rsqrtf(var + 1e-5f);
    #pragma unroll
    for (int seg = 0; seg < 6; ++seg) {
        int e = seg * H_ + tid;
        float y = (x[seg] - mean) * inv * gamma[e] + beta[e];
        Xbf[(long)r * K1_ + e] = f2bf(y);
    }
}

// ---------------- per-batch boundary scan -> scatter positions ----------------
__global__ void scan_batches(const int* __restrict__ gid, const int* __restrict__ lengths,
                             int* __restrict__ epos, int* __restrict__ spos,
                             int* __restrict__ shiftB)
{
    int b = threadIdx.x;
    if (b >= B_) return;
    const int* g = gid + (long)b * L_;
    int len = lengths[b];
    int nb = 0;
    for (int l = 0; l < L_; ++l)
        if (l < len - 1 && g[l] != g[l + 1]) ++nb;
    int shift = S_ - (len + nb);        // always >= 0
    shiftB[b] = shift;
    int ex = 0;
    for (int l = 0; l < L_; ++l) {
        bool bd = (l < len - 1) && (g[l] != g[l + 1]);
        int pe = shift + l + ex;
        long r = (long)b * L_ + l;
        epos[r] = (l < len) ? pe : -1;
        spos[r] = bd ? (pe + 1) : -1;
        ex += bd;
    }
}

// ---------------- init output: pos_table * mask, and mask itself ----------------
__global__ __launch_bounds__(256)
void init_out(const float* __restrict__ pos_table, const int* __restrict__ shiftB,
              float* __restrict__ out, float* __restrict__ mask_out)
{
    long bid = blockIdx.x;              // b*S + s
    int  b = (int)(bid / S_), s = (int)(bid % S_);
    int  tid = threadIdx.x;
    bool on = s >= shiftB[b];
    out[bid * H_ + tid] = on ? pos_table[(long)s * H_ + tid] : 0.0f;
    if (tid == 0) mask_out[bid] = on ? 1.0f : 0.0f;
}

// ---------------- bf16 WMMA GEMM: C = A(MxK) * Bt(NxK)^T + bias ----------------
// block tile 128x128, K-step 32, double-buffered LDS. 8 waves, each owns a
// 32x64 patch = 2x4 tiles of v_wmma_f32_16x16x32_bf16 (8 WMMA / wave / K-step).
// Tiles move HBM/L2 -> LDS via GLOBAL_LOAD_ASYNC_TO_LDS_B128 (ASYNCcnt), so no
// staging VGPRs exist to spill; fragments come from LDS as direct ds_load_b128.
template <int EPI>
__global__ __launch_bounds__(256, 1)
void gemm_bf16_wmma(const unsigned short* __restrict__ A,   // M x K
                    const unsigned short* __restrict__ Bt,  // N x K (transposed)
                    const float* __restrict__ bias,         // N
                    void* __restrict__ Cout,
                    int M, int N, int K)
{
    __shared__ __align__(16) unsigned short sA[2][128][48]; // 32 used + pad
    __shared__ __align__(16) unsigned short sB[2][128][48];

    const int tid  = threadIdx.x;
    const int wave = tid >> 5, lane = tid & 31;
    const int half = lane >> 4, rr = lane & 15;
    const int wm = (wave & 3) * 32;     // 4 waves down (rows)
    const int wn = (wave >> 2) * 64;    // 2 waves across (cols)
    const long m0 = (long)blockIdx.y * 128;
    const long n0 = (long)blockIdx.x * 128;

    v8f acc[2][4] = {};
    union Frag { v16bf v; uint4 q[2]; };

    auto compute = [&](int cur) {
        Frag fa[2], fb[4];
        #pragma unroll
        for (int ti = 0; ti < 2; ++ti) {
            const unsigned short* base = &sA[cur][wm + ti * 16 + rr][0];
            fa[ti].q[0] = *(const uint4*)(base + half * 8);
            fa[ti].q[1] = *(const uint4*)(base + 16 + half * 8);
        }
        #pragma unroll
        for (int tj = 0; tj < 4; ++tj) {
            const unsigned short* base = &sB[cur][wn + tj * 16 + rr][0];
            fb[tj].q[0] = *(const uint4*)(base + half * 8);
            fb[tj].q[1] = *(const uint4*)(base + 16 + half * 8);
        }
        #pragma unroll
        for (int ti = 0; ti < 2; ++ti)
            #pragma unroll
            for (int tj = 0; tj < 4; ++tj)
                acc[ti][tj] = __builtin_amdgcn_wmma_f32_16x16x32_bf16(
                    false, fa[ti].v, false, fb[tj].v,
                    (short)0, acc[ti][tj], false, false);
    };

    const int nk = K >> 5;                      // K / 32

#if HAVE_ASYNC_LDS
    // 4 async b128 loads per thread per tile (2 A-quads + 2 B-quads)
    auto issue = [&](int kt) {
        const int buf = kt & 1;
        const int k0  = kt << 5;
        #pragma unroll
        for (int t = 0; t < 2; ++t) {
            int idx = tid + t * 256;             // 0..511
            int row = idx >> 2, q = idx & 3;     // 128 rows x 4 quads
            __builtin_amdgcn_global_load_async_to_lds_b128(
                (v4i*)(A + (m0 + row) * (long)K + k0 + q * 8),
                (v4i*)&sA[buf][row][q * 8], 0, 0);
            __builtin_amdgcn_global_load_async_to_lds_b128(
                (v4i*)(Bt + (n0 + row) * (long)K + k0 + q * 8),
                (v4i*)&sB[buf][row][q * 8], 0, 0);
        }
    };

    issue(0);
    for (int kt = 0; kt < nk - 1; ++kt) {
        issue(kt + 1);                               // DMA of next tile in flight
        __builtin_amdgcn_s_wait_asynccnt(4);         // this tile's 4 loads done
        __syncthreads();                             // ... for every thread
        compute(kt & 1);
        __syncthreads();                             // reads done before tile kt+2 lands
    }
    __builtin_amdgcn_s_wait_asynccnt(0);
    __syncthreads();
    compute((nk - 1) & 1);
#else
    // fallback: register-staged copy (launch_bounds(256,1) prevents spills)
    uint4 ra[2], rb[2];
    auto gload = [&](int k0) {
        #pragma unroll
        for (int t = 0; t < 2; ++t) {
            int idx = tid + t * 256;
            int row = idx >> 2, q = idx & 3;
            ra[t] = ((const uint4*)(A  + (m0 + row) * (long)K + k0))[q];
            rb[t] = ((const uint4*)(Bt + (n0 + row) * (long)K + k0))[q];
        }
    };
    auto sstore = [&](int buf) {
        #pragma unroll
        for (int t = 0; t < 2; ++t) {
            int idx = tid + t * 256;
            int row = idx >> 2, q = idx & 3;
            *(uint4*)&sA[buf][row][q * 8] = ra[t];
            *(uint4*)&sB[buf][row][q * 8] = rb[t];
        }
    };
    gload(0); sstore(0); __syncthreads();
    for (int kt = 0; kt < nk; ++kt) {
        if (kt + 1 < nk) gload((kt + 1) << 5);
        compute(kt & 1);
        if (kt + 1 < nk) sstore((kt + 1) & 1);
        __syncthreads();
    }
#endif

    // epilogue: C/D layout — VGPR j holds row (j + 8*half), lane rr = column
    #pragma unroll
    for (int ti = 0; ti < 2; ++ti)
        #pragma unroll
        for (int tj = 0; tj < 4; ++tj)
            #pragma unroll
            for (int v = 0; v < 8; ++v) {
                long row = m0 + wm + ti * 16 + v + half * 8;
                long col = n0 + wn + tj * 16 + rr;
                float x = acc[ti][tj][v] + bias[col];
                if (EPI == 0) {
                    float sl = x / (1.0f + __expf(-x));   // SiLU
                    ((unsigned short*)Cout)[row * N + col] = f2bf(sl);
                } else {
                    ((float*)Cout)[row * N + col] = x;
                }
            }
}

// ---------------- scatter events + sep tokens into merged output ----------------
__global__ __launch_bounds__(256)
void scatter_ev(const float* __restrict__ Ev, const float* __restrict__ sep,
                const int* __restrict__ epos, const int* __restrict__ spos,
                float* __restrict__ out)
{
    long r = blockIdx.x;                 // b*L + l
    int  tid = threadIdx.x;
    int  b = (int)(r >> 9);              // /L_
    long base = (long)b * S_ * H_;
    int pe = epos[r], ps = spos[r];
    if (pe >= 0) out[base + (long)pe * H_ + tid] += Ev[r * H_ + tid];
    if (ps >= 0) out[base + (long)ps * H_ + tid] += sep[tid];
}

// ---------------- host orchestration ----------------
extern "C" void kernel_launch(void* const* d_in, const int* in_sizes, int n_in,
                              void* d_out, int out_size, void* d_ws, size_t ws_size,
                              hipStream_t stream)
{
    const int*   tok   = (const int*)  d_in[0];
    const int*   post  = (const int*)  d_in[1];
    const int*   auth  = (const int*)  d_in[2];
    const int*   act   = (const int*)  d_in[3];
    const int*   tg    = (const int*)  d_in[4];
    const int*   gid   = (const int*)  d_in[5];
    const int*   lens  = (const int*)  d_in[6];
    const float* tokT  = (const float*)d_in[7];
    const float* timT  = (const float*)d_in[8];
    const float* grpT  = (const float*)d_in[9];
    const float* posT  = (const float*)d_in[10];
    const float* sep   = (const float*)d_in[11];
    const float* gamma = (const float*)d_in[12];
    const float* beta  = (const float*)d_in[13];
    const float* W1    = (const float*)d_in[14];
    const float* b1    = (const float*)d_in[15];
    const float* W2    = (const float*)d_in[16];
    const float* b2    = (const float*)d_in[17];

    // workspace layout (bytes)
    char* ws = (char*)d_ws;
    unsigned short* Xbf  = (unsigned short*)(ws);                                 // 65536*1536 bf16
    size_t off = (size_t)NROW * K1_ * 2;
    unsigned short* Hbf  = (unsigned short*)(ws + off); off += (size_t)NROW * N1_ * 2;
    unsigned short* W1t  = (unsigned short*)(ws + off); off += (size_t)K1_ * N1_ * 2; // N1 x K1
    unsigned short* W2t  = (unsigned short*)(ws + off); off += (size_t)N1_ * H_ * 2;  // H x N1
    int* epos   = (int*)(ws + off); off += (size_t)NROW * 4;
    int* spos   = (int*)(ws + off); off += (size_t)NROW * 4;
    int* shiftB = (int*)(ws + off); off += (size_t)B_ * 4;
    float* Ev   = (float*)ws;        // aliases Xbf — Xbf is dead once GEMM1 finished

    float* merged   = (float*)d_out;                       // B*S*H
    float* mask_out = merged + (size_t)B_ * S_ * H_;       // B*S

    // 1) weights -> bf16, transposed to N-major (K-contiguous) for the GEMM
    {
        long n1 = (long)K1_ * N1_, n2 = (long)N1_ * H_;
        cvt_transpose_bf16<<<(int)((n1 + 255) / 256), 256, 0, stream>>>(W1, W1t, K1_, N1_);
        cvt_transpose_bf16<<<(int)((n2 + 255) / 256), 256, 0, stream>>>(W2, W2t, N1_, H_);
    }
    // 2) embed + LayerNorm -> X
    embed_ln<<<NROW, 256, 0, stream>>>(tok, post, auth, act, tg, gid,
                                       tokT, timT, grpT, gamma, beta, Xbf);
    // 3) boundary scan -> scatter positions
    scan_batches<<<1, B_, 0, stream>>>(gid, lens, epos, spos, shiftB);
    // 4) init merged = pos_table * mask, write mask
    init_out<<<B_ * S_, 256, 0, stream>>>(posT, shiftB, merged, mask_out);
    // 5) GEMM1: [65536x1536]x[1536x1024] + b1, SiLU -> bf16 H
    gemm_bf16_wmma<0><<<dim3(N1_ / 128, NROW / 128), 256, 0, stream>>>(
        Xbf, W1t, b1, (void*)Hbf, NROW, N1_, K1_);
    // 6) GEMM2: [65536x1024]x[1024x256] + b2 -> f32 Ev (reuses Xbf space)
    gemm_bf16_wmma<1><<<dim3(H_ / 128, NROW / 128), 256, 0, stream>>>(
        Hbf, W2t, b2, (void*)Ev, NROW, H_, N1_);
    // 7) scatter events + separators on top of pos_table
    scatter_ev<<<NROW, 256, 0, stream>>>(Ev, sep, epos, spos, merged);
}
